// LinearAttention_42537356099926
// MI455X (gfx1250) — matrix operand order repaired and checked by reading, not compiled, not run on hardware
//
#include <hip/hip_runtime.h>
#include <hip/hip_bf16.h>

// ---------------------------------------------------------------------------
// Problem constants (from the reference)
// ---------------------------------------------------------------------------
#define BATCH   2
#define SEQL    4096
#define DMODEL  1024
#define NHEADS  16
#define DK      64              // DMODEL / NHEADS
#define MROWS   (BATCH * SEQL)  // 8192
#define NCHUNK  64              // SEQL / 64 chunks for the scan
#define CHUNKL  64

typedef __attribute__((ext_vector_type(16))) _Float16 v16h;
typedef __attribute__((ext_vector_type(8)))  float    v8f;
typedef __attribute__((ext_vector_type(4)))  unsigned int v4u;   // 16 bytes
typedef __attribute__((ext_vector_type(4)))  _Float16 v4h;       // 8 bytes

union Frag16 { v16h h; v4u u[2]; };

// ---------------------------------------------------------------------------
// fp32 -> fp16 down-conversion (vectorized, 4 elems/thread)
// ---------------------------------------------------------------------------
__global__ __launch_bounds__(256)
void cvt_f32_to_f16(const float* __restrict__ in, _Float16* __restrict__ out, int n4) {
    int i = blockIdx.x * 256 + threadIdx.x;
    if (i < n4) {
        float4 v = ((const float4*)in)[i];
        v4h h = { (_Float16)v.x, (_Float16)v.y, (_Float16)v.z, (_Float16)v.w };
        ((v4h*)out)[i] = h;
    }
}

// ---------------------------------------------------------------------------
// WMMA GEMM:  out[M,N] = A[M,K] @ W[N,K]^T + bias   (torch Linear semantics)
// A, W in f16; accumulate f32 via v_wmma_f32_16x16x32_f16.
// Per wave: 32(M) x 64(N) tile -> 8 accumulators. All fragment loads for a
// K-step (2 A + 4 B) are issued up front so the in-order VMEM returns let the
// WMMA sequence use staged partial loadcnt waits instead of full drains.
// Block = 8 waves = 256(M) x 64(N).
// mode==1 applies elu(x)+1  (== x+1 if x>0 else exp(x)) after the bias.
// Fragment layouts per CDNA5 ISA 7.12.2:
//   A 16x32 f16 : lane half (lane>=16) selects K base 0/8; elems 0..7 = K
//                 base..base+7, elems 8..15 = K base+16..base+23.
//   B 32x16 f16 : same K order as A (K base 0/8; elems 8..15 = K base+16..base+23).
//   C/D 16x16   : VGPR r: lanes 0-15 M=r, lanes 16-31 M=r+8; N = lane&15.
// ---------------------------------------------------------------------------
__global__ __launch_bounds__(256)
void gemm_wmma_f16(const _Float16* __restrict__ A,     // [M, K] row major
                   const _Float16* __restrict__ W,     // [N, K] row major
                   const float*    __restrict__ bias,  // [N]
                   float*          __restrict__ out,   // [M, N]
                   int mode) {
    const int K = DMODEL;
    const int N = DMODEL;

    const int lane  = threadIdx.x & 31;
    const int wave  = threadIdx.x >> 5;
    const int half  = lane >> 4;      // 0 | 1
    const int l16   = lane & 15;

    const int mBase = blockIdx.x * 256 + wave * 32;
    const int nBase = blockIdx.y * 64;

    const _Float16* a0Ptr = A + (size_t)(mBase + l16)      * K + half * 8;
    const _Float16* a1Ptr = A + (size_t)(mBase + 16 + l16) * K + half * 8;
    const _Float16* bPtr  = W + (size_t)(nBase + l16) * K + half * 8;

    v8f acc[8] = {v8f{}, v8f{}, v8f{}, v8f{}, v8f{}, v8f{}, v8f{}, v8f{}};

    #pragma unroll 2
    for (int k0 = 0; k0 < K; k0 += 32) {
        // --- issue ALL fragment loads for this K-step first (one clause) ---
        Frag16 a0, a1, b[4];
        const v4u* ap0 = (const v4u*)(a0Ptr + k0);
        const v4u* ap1 = (const v4u*)(a1Ptr + k0);
        a0.u[0] = ap0[0];  a0.u[1] = ap0[2];   // K = base..+7, base+16..+23
        a1.u[0] = ap1[0];  a1.u[1] = ap1[2];
        #pragma unroll
        for (int j = 0; j < 4; ++j) {
            const v4u* bp = (const v4u*)(bPtr + (size_t)j * 16 * K + k0);
            b[j].u[0] = bp[0];
            b[j].u[1] = bp[2];
        }
        if (k0 + 64 < K) {                      // global_prefetch_b8 next K-slab
            __builtin_prefetch((const void*)(a0Ptr + k0 + 64), 0, 3);
            __builtin_prefetch((const void*)(a1Ptr + k0 + 64), 0, 3);
        }

        // --- 8 WMMAs; later B fragments may still be in flight ---
        #pragma unroll
        for (int j = 0; j < 4; ++j) {
            acc[2 * j] = __builtin_amdgcn_wmma_f32_16x16x32_f16(
                             false, a0.h, false, b[j].h, (short)0, acc[2 * j],
                             false, false);
            acc[2 * j + 1] = __builtin_amdgcn_wmma_f32_16x16x32_f16(
                             false, a1.h, false, b[j].h, (short)0, acc[2 * j + 1],
                             false, false);
        }
    }

    // Epilogue: bias (+ optional elu+1), write f32.
    #pragma unroll
    for (int j = 0; j < 4; ++j) {
        const int n  = nBase + j * 16 + l16;
        const float bv = bias[n];
        #pragma unroll
        for (int sub = 0; sub < 2; ++sub) {
            #pragma unroll
            for (int r = 0; r < 8; ++r) {
                const int m = mBase + sub * 16 + half * 8 + r;
                float x = acc[2 * j + sub][r] + bv;
                if (mode) x = (x > 0.0f) ? (x + 1.0f) : __expf(x);  // elu(x)+1
                out[(size_t)m * N + n] = x;
            }
        }
    }
}

// ---------------------------------------------------------------------------
// Scan pass A: per (b,h,chunk) 64-step sums of kf and vf over L.
// Block = 1 wave (32 lanes); lane t owns head dims t and t+32.
// ---------------------------------------------------------------------------
__global__ __launch_bounds__(32)
void chunk_sums(const float* __restrict__ kf, const float* __restrict__ vf,
                float* __restrict__ ksumC, float* __restrict__ vsumC) {
    const int t  = threadIdx.x;
    const int c  = blockIdx.x % NCHUNK;
    const int bh = blockIdx.x / NCHUNK;
    const int h  = bh % NHEADS;
    const int b  = bh / NHEADS;

    const size_t base = ((size_t)b * SEQL + (size_t)c * CHUNKL) * DMODEL + (size_t)h * DK;
    float ks0 = 0.f, ks1 = 0.f, vs0 = 0.f, vs1 = 0.f;
    for (int l = 0; l < CHUNKL; ++l) {
        const size_t off = base + (size_t)l * DMODEL;
        ks0 += kf[off + t];      ks1 += kf[off + t + 32];
        vs0 += vf[off + t];      vs1 += vf[off + t + 32];
    }
    const size_t o = ((size_t)bh * NCHUNK + c) * DK;
    ksumC[o + t] = ks0;  ksumC[o + t + 32] = ks1;
    vsumC[o + t] = vs0;  vsumC[o + t + 32] = vs1;
}

// ---------------------------------------------------------------------------
// Scan pass B: replay each chunk with the exclusive chunk-prefix carry.
// Computes z from the full-sequence k-sum, the inclusive cumsum carries,
// the per-token qf.k_cum dot (wave32 shfl reduction) and writes the scaled
// context directly as f16 (feeds the final WMMA GEMM).
// ---------------------------------------------------------------------------
__global__ __launch_bounds__(32)
void scan_ctx(const float* __restrict__ qf, const float* __restrict__ kf,
              const float* __restrict__ vf,
              const float* __restrict__ ksumC, const float* __restrict__ vsumC,
              _Float16* __restrict__ ctx) {
    const int t  = threadIdx.x;
    const int c  = blockIdx.x % NCHUNK;
    const int bh = blockIdx.x / NCHUNK;
    const int h  = bh % NHEADS;
    const int b  = bh / NHEADS;

    // Chunk totals (all chunks -> z) and exclusive prefix (chunks < c -> carry)
    float kTot0 = 0.f, kTot1 = 0.f;
    float kRun0 = 0.f, kRun1 = 0.f, vRun0 = 0.f, vRun1 = 0.f;
    const size_t sbase = (size_t)bh * NCHUNK * DK;
    for (int cc = 0; cc < NCHUNK; ++cc) {
        const size_t so = sbase + (size_t)cc * DK;
        const float a0 = ksumC[so + t], a1 = ksumC[so + t + 32];
        kTot0 += a0;  kTot1 += a1;
        if (cc < c) {
            kRun0 += a0;  kRun1 += a1;
            vRun0 += vsumC[so + t];  vRun1 += vsumC[so + t + 32];
        }
    }

    const size_t base = ((size_t)b * SEQL + (size_t)c * CHUNKL) * DMODEL + (size_t)h * DK;
    for (int l = 0; l < CHUNKL; ++l) {
        const size_t off = base + (size_t)l * DMODEL;
        const float q0 = qf[off + t], q1 = qf[off + t + 32];
        kRun0 += kf[off + t];  kRun1 += kf[off + t + 32];   // inclusive cumsum
        vRun0 += vf[off + t];  vRun1 += vf[off + t + 32];

        float s  = q0 * kRun0 + q1 * kRun1;   // qf . k_cum   (partial)
        float zz = q0 * kTot0 + q1 * kTot1;   // qf . k_total (partial)
        #pragma unroll
        for (int m = 16; m > 0; m >>= 1) {    // wave32 reduction
            s  += __shfl_xor(s,  m, 32);
            zz += __shfl_xor(zz, m, 32);
        }
        const float scale = s * (1.0f / (zz + 1e-6f));
        ctx[off + t]      = (_Float16)(scale * vRun0);
        ctx[off + t + 32] = (_Float16)(scale * vRun1);
    }
}

// ---------------------------------------------------------------------------
// Host-side pipeline (all launches on `stream`, graph-capture safe)
// ---------------------------------------------------------------------------
extern "C" void kernel_launch(void* const* d_in, const int* in_sizes, int n_in,
                              void* d_out, int out_size, void* d_ws, size_t ws_size,
                              hipStream_t stream) {
    const float* q  = (const float*)d_in[0];
    const float* k  = (const float*)d_in[1];
    const float* v  = (const float*)d_in[2];
    const float* Wq = (const float*)d_in[3];
    const float* bq = (const float*)d_in[4];
    const float* Wk = (const float*)d_in[5];
    const float* bk = (const float*)d_in[6];
    const float* Wv = (const float*)d_in[7];
    const float* bv = (const float*)d_in[8];
    const float* Wo = (const float*)d_in[9];
    const float* bo = (const float*)d_in[10];
    float* out = (float*)d_out;

    // Workspace carve-up (~115 MB)
    char* ws = (char*)d_ws;
    _Float16* hX = (_Float16*)(ws);                               // 16 MB f16 staging
    _Float16* hW = (_Float16*)(ws + ((size_t)16 << 20));          //  2 MB f16 weights
    float* qf    = (float*)(ws + ((size_t)18 << 20));             // 32 MB
    float* kf    = (float*)(ws + ((size_t)50 << 20));             // 32 MB
    float* vf    = (float*)(ws + ((size_t)82 << 20));             // 32 MB
    float* ksumC = (float*)(ws + ((size_t)114 << 20));            // 0.5 MB
    float* vsumC = ksumC + (size_t)BATCH * NHEADS * NCHUNK * DK;  // 0.5 MB

    const int nX4 = (MROWS * DMODEL) / 4;
    const int nW4 = (DMODEL * DMODEL) / 4;
    dim3 cb(256);
    dim3 gX((nX4 + 255) / 256), gW((nW4 + 255) / 256);
    dim3 gg(MROWS / 256, DMODEL / 64), gb(256);
    dim3 gs(BATCH * NHEADS * NCHUNK), sb(32);

    // Q projection (elu+1)
    cvt_f32_to_f16<<<gX, cb, 0, stream>>>(q,  hX, nX4);
    cvt_f32_to_f16<<<gW, cb, 0, stream>>>(Wq, hW, nW4);
    gemm_wmma_f16 <<<gg, gb, 0, stream>>>(hX, hW, bq, qf, 1);
    // K projection (elu+1)
    cvt_f32_to_f16<<<gX, cb, 0, stream>>>(k,  hX, nX4);
    cvt_f32_to_f16<<<gW, cb, 0, stream>>>(Wk, hW, nW4);
    gemm_wmma_f16 <<<gg, gb, 0, stream>>>(hX, hW, bk, kf, 1);
    // V projection (plain)
    cvt_f32_to_f16<<<gX, cb, 0, stream>>>(v,  hX, nX4);
    cvt_f32_to_f16<<<gW, cb, 0, stream>>>(Wv, hW, nW4);
    gemm_wmma_f16 <<<gg, gb, 0, stream>>>(hX, hW, bv, vf, 0);

    // Chunked linear-attention scan -> f16 context into hX
    chunk_sums<<<gs, sb, 0, stream>>>(kf, vf, ksumC, vsumC);
    scan_ctx  <<<gs, sb, 0, stream>>>(qf, kf, vf, ksumC, vsumC, hX);

    // Output projection -> d_out (f32)
    cvt_f32_to_f16<<<gW, cb, 0, stream>>>(Wo, hW, nW4);
    gemm_wmma_f16 <<<gg, gb, 0, stream>>>(hX, hW, bo, out, 0);
}
